// PairwiseMLPSimilarityLoop_3848290697818
// MI455X (gfx1250) — matrix operand order repaired and checked by reading, not compiled, not run on hardware
//
#include <hip/hip_runtime.h>
#include <hip/hip_bf16.h>
#include <cstdint>

typedef __attribute__((ext_vector_type(16))) _Float16 v16h;
typedef __attribute__((ext_vector_type(8)))  float    v8f;

#define Bsz 4
#define Nn  1024
#define Ee  32
#define Hh  64

// ---------------------------------------------------------------------------
// Stage 1: per 16-row tile of flat X [B*N, E]:
//   Aws  = X @ W1c[0:32,:]  + b1c      (pair MLP, "i" half, bias folded in)
//   Bws  = X @ W1c[32:64,:]            (pair MLP, "j" half)
//   S    = sigmoid( relu(X @ W1s + b1s) @ W2s + b2s )   (diagonal)
// One wave (32 threads) per block; v_wmma_f32_16x16x32_f16, reusing one
// A-fragment across all 12 tiles.
// ---------------------------------------------------------------------------
__global__ __launch_bounds__(32) void pmlp_precompute_kernel(
    const float* __restrict__ X,   const float* __restrict__ W1c,
    const float* __restrict__ b1c, const float* __restrict__ W1s,
    const float* __restrict__ b1s, const float* __restrict__ W2s,
    const float* __restrict__ b2s,
    float* __restrict__ Aws, float* __restrict__ Bws, float* __restrict__ S)
{
    __shared__ float sH[16][Hh];

    const int lane = threadIdx.x;        // 0..31
    const int m    = lane & 15;          // M index within tile
    const int hi   = lane >> 4;          // lane-half selector
    const int row0 = blockIdx.x * 16;    // 16 rows of flat [B*N]

    // A-fragment: 16x32 f16, layout per CDNA5 ISA 7.12.2:
    // element idx -> K = 16*(idx>>3) + 8*hi + (idx&7), row = m
    v16h a;
    #pragma unroll
    for (int idx = 0; idx < 16; ++idx) {
        const int k = 16 * (idx >> 3) + 8 * hi + (idx & 7);
        a[idx] = (_Float16)X[(size_t)(row0 + m) * Ee + k];
    }

    #pragma unroll 1
    for (int sel = 0; sel < 3; ++sel) {
        // sel 0: top half of W1c; sel 1: bottom half; sel 2: W1s
        const float* W = (sel == 0) ? W1c : (sel == 1) ? (W1c + Ee * Hh) : W1s;
        #pragma unroll 1
        for (int ct = 0; ct < 4; ++ct) {
            const int n = ct * 16 + m;   // output column (0..63)

            // B-fragment: 32x16 f16, column-major stripe:
            // element idx -> K = 16*(idx>>3) + 8*hi + (idx&7)
            v16h bf;
            #pragma unroll
            for (int idx = 0; idx < 16; ++idx) {
                const int k = 16 * (idx >> 3) + 8 * hi + (idx & 7);
                bf[idx] = (_Float16)W[(size_t)k * Hh + n];
            }

            const float binit = (sel == 0) ? b1c[n] : (sel == 2) ? b1s[n] : 0.0f;
            v8f acc;
            #pragma unroll
            for (int r = 0; r < 8; ++r) acc[r] = binit;

            acc = __builtin_amdgcn_wmma_f32_16x16x32_f16(
                      false, a, false, bf, (short)0, acc, false, false);

            if (sel < 2) {
                float* Out = sel ? Bws : Aws;
                // C/D layout: VGPR r -> row (r + 8*hi), col n
                #pragma unroll
                for (int r = 0; r < 8; ++r)
                    Out[(size_t)(row0 + r + 8 * hi) * Hh + n] = acc[r];
            } else {
                #pragma unroll
                for (int r = 0; r < 8; ++r)
                    sH[r + 8 * hi][n] = fmaxf(acc[r], 0.0f) * W2s[n];
            }
        }
    }
    __syncthreads();

    if (lane < 16) {
        float s = b2s[0];
        #pragma unroll
        for (int k = 0; k < Hh; ++k) s += sH[lane][k];
        S[row0 + lane] = 1.0f / (1.0f + __expf(-s));
    }
}

// ---------------------------------------------------------------------------
// Stage 2: sim[b,i,j]. Upper-triangular 16x16 tiles only; mirrored store.
// A/B row tiles are contiguous 4 KB blocks in the workspace -> fetched with
// GLOBAL_LOAD_ASYNC_TO_LDS_B128 (one b128 per lane, ASYNCcnt-tracked), then
// consumed as float4 LDS reads (ds_load_b128).
// ---------------------------------------------------------------------------
__global__ __launch_bounds__(256) void pmlp_pairwise_kernel(
    const float* __restrict__ Aws, const float* __restrict__ Bws,
    const float* __restrict__ S,   const float* __restrict__ W2c,
    const float* __restrict__ b2c, float* __restrict__ out)
{
    __shared__ __align__(16) float sA[16][Hh];
    __shared__ __align__(16) float sB[16][Hh];
    __shared__ __align__(16) float sW[Hh];

    const int b  = blockIdx.z;
    const int i0 = blockIdx.y * 16;
    const int j0 = blockIdx.x * 16;
    if (j0 < i0) return;                 // symmetry: skip lower tiles

    const int t = threadIdx.x;           // 0..255
    const size_t base = (size_t)b * Nn;

    // Async copy global -> LDS: 256 lanes x 16B covers each 4 KB tile.
    {
        unsigned lA = (unsigned)(uintptr_t)(&sA[0][0]) + (unsigned)t * 16u;
        unsigned lB = (unsigned)(uintptr_t)(&sB[0][0]) + (unsigned)t * 16u;
        const char* gA = (const char*)(Aws + (base + i0) * Hh) + t * 16;
        const char* gB = (const char*)(Bws + (base + j0) * Hh) + t * 16;
        asm volatile("global_load_async_to_lds_b128 %0, %1, off"
                     :: "v"(lA), "v"(gA) : "memory");
        asm volatile("global_load_async_to_lds_b128 %0, %1, off"
                     :: "v"(lB), "v"(gB) : "memory");
    }
    if (t < Hh) sW[t] = W2c[t];
    asm volatile("s_wait_asynccnt 0x0" ::: "memory");
    __syncthreads();

    const int tx = t & 15;               // j offset in tile
    const int ty = t >> 4;               // i offset in tile
    const int i = i0 + ty, j = j0 + tx;

    float val;
    if (i == j) {
        val = S[base + i];
    } else {
        // ordered pair: A row from min(i,j), Bm row from max(i,j)
        const float4* ar = (const float4*)((i < j) ? sA[ty] : sA[tx]);
        const float4* br = (const float4*)((i < j) ? sB[tx] : sB[ty]);
        const float4* wr = (const float4*)sW;
        float acc = b2c[0];
        #pragma unroll
        for (int k = 0; k < Hh / 4; ++k) {
            const float4 av = ar[k], bv = br[k], wv = wr[k];
            acc += fmaxf(av.x + bv.x, 0.0f) * wv.x;
            acc += fmaxf(av.y + bv.y, 0.0f) * wv.y;
            acc += fmaxf(av.z + bv.z, 0.0f) * wv.z;
            acc += fmaxf(av.w + bv.w, 0.0f) * wv.w;
        }
        val = 1.0f / (1.0f + __expf(-acc));
    }

    out[((size_t)b * Nn + i) * Nn + j] = val;
    if (i0 != j0)
        out[((size_t)b * Nn + j) * Nn + i] = val;
}

// ---------------------------------------------------------------------------
extern "C" void kernel_launch(void* const* d_in, const int* in_sizes, int n_in,
                              void* d_out, int out_size, void* d_ws, size_t ws_size,
                              hipStream_t stream) {
    const float* X   = (const float*)d_in[0];
    const float* W1c = (const float*)d_in[1];
    const float* b1c = (const float*)d_in[2];
    const float* W2c = (const float*)d_in[3];
    const float* b2c = (const float*)d_in[4];
    const float* W1s = (const float*)d_in[5];
    const float* b1s = (const float*)d_in[6];
    const float* W2s = (const float*)d_in[7];
    const float* b2s = (const float*)d_in[8];
    float* out = (float*)d_out;

    // workspace: A [4096,64] | Bm [4096,64] | S [4096]  (~2.02 MB)
    float* Aws = (float*)d_ws;
    float* Bws = Aws + (size_t)Bsz * Nn * Hh;
    float* S   = Bws + (size_t)Bsz * Nn * Hh;

    pmlp_precompute_kernel<<<(Bsz * Nn) / 16, 32, 0, stream>>>(
        X, W1c, b1c, W1s, b1s, W2s, b2s, Aws, Bws, S);

    pmlp_pairwise_kernel<<<dim3(Nn / 16, Nn / 16, Bsz), 256, 0, stream>>>(
        Aws, Bws, S, W2c, b2c, out);
}
